// Encoder_search_47665547051210
// MI455X (gfx1250) — matrix operand order repaired and checked by reading, not compiled, not run on hardware
//
#include <hip/hip_runtime.h>

// ---------------------------------------------------------------------------
// MI455X (gfx1250). 520 channel-mixing 3-tap convs as implicit GEMM on
// v_wmma_f32_16x16x32_bf16. Weight stream (~73 MB f32) is the roofline
// limiter (~3.1 us @ 23.3 TB/s HBM). Activations live in L2 and are stored
// NHWC **bf16** between kernels so the WMMA loop has zero conversion VALU:
//   A fragment: 2x ds_load_b128 (weight tile staged in LDS, converted once
//               per block and shared by 8 waves)
//   B fragment: 2x global_load_b128 (contiguous bf16 channels; OOB taps
//               redirected to a zeroed page -> no mask math, no divergence)
// ---------------------------------------------------------------------------

typedef __attribute__((ext_vector_type(16))) __bf16 v16bf;
typedef __attribute__((ext_vector_type(8)))  __bf16 v8bf;
typedef __attribute__((ext_vector_type(8)))  float  v8f;

struct BfPair { v8bf x, y; };                 // 32B -> bit_cast to v16bf

__device__ __forceinline__ __bf16 f2bf(float f) {   // nearest (ties away)
  unsigned u = __builtin_bit_cast(unsigned, f) + 0x8000u;
  return __builtin_bit_cast(__bf16, (unsigned short)(u >> 16));
}
__device__ __forceinline__ float bf2f(__bf16 h) {
  unsigned u = ((unsigned)__builtin_bit_cast(unsigned short, h)) << 16;
  return __builtin_bit_cast(float, u);
}
__device__ __forceinline__ unsigned pack2bf(float f0, float f1) {
  unsigned u0 = __builtin_bit_cast(unsigned, f0) + 0x8000u;
  unsigned u1 = __builtin_bit_cast(unsigned, f1) + 0x8000u;
  return __builtin_amdgcn_perm(u1, u0, 0x07060302u);  // {f1.hi16, f0.hi16}
}

// ---------------------------------------------------------------------------
// GEMM-formulated 3-tap separable conv, NHWC bf16 activations:
//   out[pix,co] = sum_ci sum_t w[co,ci,t] * in[pix + (t-1)*(dy,dx), ci]
// Block: 8 waves sharing one 16-channel output tile (LDS weight tile),
// each wave owns one 16-pixel group. K chunks of 32 channels share a tap.
// mode 0: outBf = relu(acc+b)
// mode 1: outBf = relu(acc+b) * gate[co]
// mode 2: outF += wo * gate[co] * (acc+b)     (f32 accumulate)
// ---------------------------------------------------------------------------
template <int C, int H, int W>
__global__ void __launch_bounds__(256)
conv3tap_wmma(const __bf16* __restrict__ in, const float* __restrict__ wgt,
              const float* __restrict__ bias, __bf16* __restrict__ outBf,
              float* __restrict__ outF, const float* __restrict__ gate,
              const float* __restrict__ wop, const __bf16* __restrict__ zpage,
              int dy, int dx, int mode)
{
  constexpr int HW      = H * W;
  constexpr int nTilesN = (2 * HW) / 16;    // 256 (64-stage) / 64 (128-stage)
  constexpr int NTB     = nTilesN / 8;      // blocks per co tile (8 pixel-groups each)

  __shared__ __bf16 wlds[16 * 3 * C];       // [tap][co_local][ci], 6/12 KB

  const int mt     = blockIdx.x / NTB;
  const int ntg    = blockIdx.x - mt * NTB;
  const int coBase = mt << 4;

  // Cooperative weight tile: stream f32 coalesced, convert once, share 8x.
  {
    const float* __restrict__ wsrc = wgt + coBase * C * 3;
    __builtin_prefetch(wsrc, 0, 3);
    for (int s = threadIdx.x; s < 16 * C * 3; s += 256) {
      const int mm = s / (C * 3);
      const int r  = s - mm * (C * 3);
      const int ci = r / 3;
      const int tt = r - ci * 3;
      wlds[(tt * 16 + mm) * C + ci] = f2bf(wsrc[s]);
    }
  }
  __syncthreads();

  const int w8      = threadIdx.x >> 5;
  const int lane    = threadIdx.x & 31;
  const int nt      = ntg * 8 + w8;
  const int posBase = nt << 4;
  const int nb      = posBase / HW;
  const int rem     = posBase - nb * HW;
  const int y0      = rem / W;
  const int x0      = rem - y0 * W;         // 16 cols share one row (16 | W)
  const int m       = lane & 15;            // A row (co) / B col (pixel)
  const int hh      = lane >> 4;

  // Per-tap B base pointer; OOB -> zero page (no mask math in the loop).
  const __bf16* bbase[3];
#pragma unroll
  for (int t = 0; t < 3; ++t) {
    const int sy  = y0 + (t - 1) * dy;
    const int xx  = x0 + m + (t - 1) * dx;
    const bool ok = (sy >= 0) && (sy < H) && (xx >= 0) && (xx < W);
    const int syc = min(max(sy, 0), H - 1);
    const int xxc = min(max(xx, 0), W - 1);
    bbase[t] = ok ? (in + (((nb * H + syc) * W + xxc) * C + (hh << 4))) : zpage;
  }

  v8f acc;
#pragma unroll
  for (int r = 0; r < 8; ++r) acc[r] = 0.0f;

#pragma unroll
  for (int kb = 0; kb < C; kb += 32) {
#pragma unroll
    for (int t = 0; t < 3; ++t) {
      // A k order chosen here (B matches; not the ISA table): k = h*16 + e -> 16 consecutive channels
      const __bf16* ap = &wlds[(t * 16 + m) * C + kb + (hh << 4)];
      BfPair A2;  A2.x = *(const v8bf*)ap;       A2.y = *(const v8bf*)(ap + 8);
      const __bf16* bp = bbase[t] + kb;          // 16 consecutive channels
      BfPair B2;  B2.x = *(const v8bf*)bp;       B2.y = *(const v8bf*)(bp + 8);
      const v16bf A = __builtin_bit_cast(v16bf, A2);
      const v16bf B = __builtin_bit_cast(v16bf, B2);
      acc = __builtin_amdgcn_wmma_f32_16x16x32_bf16(false, A, false, B,
                                                    (short)0, acc, false, false);
    }
  }

  // Epilogue: D VGPR r -> co = coBase + 8*hh + r (8 consecutive channels)
  const int cofs  = coBase + (hh << 3);
  const int obase = (posBase + m) * C + cofs;
  float v[8], bb[8];
  *(float4*)(bb + 0) = *(const float4*)(bias + cofs);
  *(float4*)(bb + 4) = *(const float4*)(bias + cofs + 4);
#pragma unroll
  for (int r = 0; r < 8; ++r) v[r] = acc[r] + bb[r];

  if (mode == 2) {
    const float wo = wop[0];
    float gg[8], oo[8];
    *(float4*)(gg + 0) = *(const float4*)(gate + cofs);
    *(float4*)(gg + 4) = *(const float4*)(gate + cofs + 4);
    *(float4*)(oo + 0) = *(const float4*)(outF + obase);
    *(float4*)(oo + 4) = *(const float4*)(outF + obase + 4);
#pragma unroll
    for (int r = 0; r < 8; ++r) v[r] = oo[r] + wo * gg[r] * v[r];
    *(float4*)(outF + obase)     = *(float4*)(v + 0);
    *(float4*)(outF + obase + 4) = *(float4*)(v + 4);
  } else {
    if (mode == 0) {
#pragma unroll
      for (int r = 0; r < 8; ++r) v[r] = fmaxf(v[r], 0.0f);
    } else {
      float gg[8];
      *(float4*)(gg + 0) = *(const float4*)(gate + cofs);
      *(float4*)(gg + 4) = *(const float4*)(gate + cofs + 4);
#pragma unroll
      for (int r = 0; r < 8; ++r) v[r] = fmaxf(v[r], 0.0f) * gg[r];
    }
    uint4 pk;
    pk.x = pack2bf(v[0], v[1]);  pk.y = pack2bf(v[2], v[3]);
    pk.z = pack2bf(v[4], v[5]);  pk.w = pack2bf(v[6], v[7]);
    *(uint4*)(outBf + obase) = pk;             // one b128 store
  }
}

// ---------------------------------------------------------------------------
// Downsampler: conv3x3 s2 p1 (Cin->Cconv) || maxpool2x2 (Cin), concat,
// eval-BN + ReLU. Output NHWC bf16. inF (NCHW f32, ds0) or inB (NHWC bf16).
// ---------------------------------------------------------------------------
__global__ void downsampler_k(const float* __restrict__ inF, const __bf16* __restrict__ inB,
                              const float* __restrict__ cw, const float* __restrict__ cb,
                              const float* __restrict__ g, const float* __restrict__ bt,
                              __bf16* __restrict__ out,
                              int Cin, int Cconv, int Hin, int Win)
{
  const int Ho = Hin >> 1, Wo = Win >> 1, Cout = Cconv + Cin;
  const int idx = blockIdx.x * blockDim.x + threadIdx.x;
  if (idx >= 2 * Cout * Ho * Wo) return;
  const int co = idx % Cout;                   // NHWC: channel innermost
  int t        = idx / Cout;
  const int x  = t % Wo;  t /= Wo;
  const int y  = t % Ho;
  const int nb = t / Ho;
  float v;
  if (co < Cconv) {
    float s = cb[co];
    for (int ci = 0; ci < Cin; ++ci)
      for (int ky = 0; ky < 3; ++ky) {
        const int iy = 2 * y + ky - 1;
        if (iy < 0 || iy >= Hin) continue;
        for (int kx = 0; kx < 3; ++kx) {
          const int ix = 2 * x + kx - 1;
          if (ix < 0 || ix >= Win) continue;
          const float xv = inF ? inF[((nb * Cin + ci) * Hin + iy) * Win + ix]
                               : bf2f(inB[((nb * Hin + iy) * Win + ix) * Cin + ci]);
          s += xv * cw[((co * Cin + ci) * 3 + ky) * 3 + kx];
        }
      }
    v = s;
  } else {
    const int ci = co - Cconv;
    float p00, p01, p10, p11;
    if (inF) {
      const float* p = inF + ((nb * Cin + ci) * Hin + 2 * y) * Win + 2 * x;
      p00 = p[0]; p01 = p[1]; p10 = p[Win]; p11 = p[Win + 1];
    } else {
      const __bf16* p = inB + ((nb * Hin + 2 * y) * Win + 2 * x) * Cin + ci;
      p00 = bf2f(p[0]); p01 = bf2f(p[Cin]);
      p10 = bf2f(p[Win * Cin]); p11 = bf2f(p[Win * Cin + Cin]);
    }
    v = fmaxf(fmaxf(p00, p01), fmaxf(p10, p11));
  }
  v = v * (g[co] * 0.99950037f) + bt[co];      // 1/sqrt(1.001)
  out[idx] = f2bf(fmaxf(v, 0.0f));
}

// Routing: wo[13][10] = norm(sigmoid(alphas)*switch)
__global__ void route_wo_k(const float* __restrict__ a, const unsigned char* __restrict__ sw,
                           float* __restrict__ wo)
{
  __shared__ float s[10];
  const int l = blockIdx.x, j = threadIdx.x;
  if (j < 10) {
    const float v = 1.0f / (1.0f + expf(-a[l * 10 + j]));
    s[j] = sw[l * 10 + j] ? v : 0.0f;
  }
  __syncthreads();
  if (j < 10) {
    float sum = 0.0f;
    for (int k = 0; k < 10; ++k) sum += s[k];
    wo[l * 10 + j] = s[j] / sum;
  }
}

// Per (layer,primitive): normalized woc/wocc folded into per-channel gates
// gate[c] = sum_j w[j]*(c < C - C*j/8), plus channel sums for the FLOPs chain.
__global__ void route_gates_k(const float* __restrict__ a1, const unsigned char* __restrict__ s1,
                              const float* __restrict__ a2, const unsigned char* __restrict__ s2,
                              float* __restrict__ lstgate, float* __restrict__ midgate,
                              float* __restrict__ lstsum, float* __restrict__ midsum)
{
  __shared__ float wc[9], wcc[9], redL[128], redM[128];
  const int li = blockIdx.x;                 // 0..129 = l*10+i
  const int C  = ((li / 10) < 5) ? 64 : 128;
  const int t  = threadIdx.x;
  if (t < 9) {
    const float v1 = 1.0f / (1.0f + expf(-a1[li * 9 + t]));
    const float v2 = 1.0f / (1.0f + expf(-a2[li * 9 + t]));
    wc[t]  = s1[li * 9 + t] ? v1 : 0.0f;
    wcc[t] = s2[li * 9 + t] ? v2 : 0.0f;
  }
  __syncthreads();
  if (t == 0) {
    float u1 = 0.0f, u2 = 0.0f;
    for (int j = 0; j < 9; ++j) { u1 += wc[j]; u2 += wcc[j]; }
    for (int j = 0; j < 9; ++j) { wc[j] /= u1; wcc[j] /= u2; }
  }
  __syncthreads();
  float gl = 0.0f, gm = 0.0f;
  if (t < C)
    for (int j = 0; j < 9; ++j) {
      const int keep = C - (C * j) / 8;
      if (t < keep) { gl += wc[j]; gm += wcc[j]; }
    }
  lstgate[li * 128 + t] = (t < C) ? gl : 0.0f;
  midgate[li * 128 + t] = (t < C) ? gm : 0.0f;
  redL[t] = (t < C) ? gl : 0.0f;
  redM[t] = (t < C) ? gm : 0.0f;
  __syncthreads();
  for (int s = 64; s > 0; s >>= 1) {
    if (t < s) { redL[t] += redL[t + s]; redM[t] += redM[t + s]; }
    __syncthreads();
  }
  if (t == 0) { lstsum[li] = redL[0]; midsum[li] = redM[0]; }
}

__global__ void zero_k(float* __restrict__ p, int n) {
  const int i = blockIdx.x * blockDim.x + threadIdx.x;
  if (i < n) p[i] = 0.0f;
}
// relu(f32 accum) -> bf16 next-layer input
__global__ void reluconv_k(const float* __restrict__ src, __bf16* __restrict__ dst, int n) {
  const int i = blockIdx.x * blockDim.x + threadIdx.x;
  if (i < n) dst[i] = f2bf(fmaxf(src[i], 0.0f));
}

// 1x1 conv 128 -> 20; input NHWC bf16 [2,16,32,128], output NCHW f32 [2,20,16,32]
__global__ void outconv_k(const __bf16* __restrict__ in, const float* __restrict__ w,
                          const float* __restrict__ b, float* __restrict__ out)
{
  const int idx = blockIdx.x * blockDim.x + threadIdx.x;
  if (idx >= 2 * 20 * 512) return;
  const int pix = idx % 512;
  const int co  = (idx / 512) % 20;
  const int nb  = idx / (512 * 20);
  const __bf16* __restrict__ ip = in + (nb * 512 + pix) * 128;
  const float*  __restrict__ wp = w + co * 128;
  float s = b[co];
  for (int ci = 0; ci < 128; ++ci) s += bf2f(ip[ci]) * wp[ci];
  out[idx] = s;
}

// Sequentially-chained FLOPs scalar (in_ch chains through layers).
__global__ void flops_k(const float* __restrict__ wo, const float* __restrict__ midsum,
                        const float* __restrict__ lstsum, float* __restrict__ dst)
{
  float flops = ((3.f*3.f*3.f+1.f)*(16.f-3.f) + 2.f*2.f*3.f) * (64.f*128.f);
  flops += ((3.f*3.f*16.f+1.f)*(64.f-16.f) + 2.f*2.f*16.f) * (32.f*64.f);
  float in_ch = 64.f;
  for (int l = 0; l < 5; ++l) {
    float lst = in_ch;
    for (int i = 0; i < 10; ++i) {
      const int li = l*10+i;
      const float mid = midsum[li], ls = lstsum[li];
      flops += wo[li] * 2048.f * ((3.f*in_ch+1.f)*mid + 2.f*(3.f*mid+1.f)*mid + (3.f*mid+1.f)*ls);
      lst = ls;
    }
    in_ch = lst;
  }
  flops += ((3.f*3.f*in_ch+1.f)*(128.f-in_ch) + 2.f*2.f*in_ch) * (16.f*32.f);
  in_ch = 128.f;
  for (int l = 5; l < 13; ++l) {
    float lst = in_ch;
    for (int i = 0; i < 10; ++i) {
      const int li = l*10+i;
      const float mid = midsum[li], ls = lstsum[li];
      flops += wo[li] * 512.f * ((3.f*in_ch+1.f)*mid + 2.f*(3.f*mid+1.f)*mid + (3.f*mid+1.f)*ls);
      lst = ls;
    }
    in_ch = lst;
  }
  flops += (128.f + 1.f) * 20.f * (16.f*32.f);
  dst[0] = flops * 1e-9f;
}

// ---------------------------------------------------------------------------
extern "C" void kernel_launch(void* const* d_in, const int* in_sizes, int n_in,
                              void* d_out, int out_size, void* d_ws, size_t ws_size,
                              hipStream_t stream)
{
  (void)in_sizes; (void)n_in; (void)out_size; (void)ws_size;

  const float* x       = (const float*)d_in[0];
  const float* alphas  = (const float*)d_in[1];
  const float* alphas1 = (const float*)d_in[2];
  const float* alphas2 = (const float*)d_in[3];
  const unsigned char* sw  = (const unsigned char*)d_in[4];   // numpy bool = 1 byte
  const unsigned char* sw1 = (const unsigned char*)d_in[5];
  const unsigned char* sw2 = (const unsigned char*)d_in[6];
  const float* init_w = (const float*)d_in[7];
  const float* init_b = (const float*)d_in[8];
  const float* bn0_g  = (const float*)d_in[9];
  const float* bn0_b  = (const float*)d_in[10];
  const float* ds1_w  = (const float*)d_in[11];
  const float* ds1_b  = (const float*)d_in[12];
  const float* bn1_g  = (const float*)d_in[13];
  const float* bn1_b  = (const float*)d_in[14];
  const float* ds2_w  = (const float*)d_in[15];
  const float* ds2_b  = (const float*)d_in[16];
  const float* bn2_g  = (const float*)d_in[17];
  const float* bn2_b  = (const float*)d_in[18];
  // setup_inputs() dict order: per i in 1..4 -> g64_w, g64_b, g128_w, g128_b
  const float* g64w[4]  = {(const float*)d_in[19], (const float*)d_in[23],
                           (const float*)d_in[27], (const float*)d_in[31]};
  const float* g64b[4]  = {(const float*)d_in[20], (const float*)d_in[24],
                           (const float*)d_in[28], (const float*)d_in[32]};
  const float* g128w[4] = {(const float*)d_in[21], (const float*)d_in[25],
                           (const float*)d_in[29], (const float*)d_in[33]};
  const float* g128b[4] = {(const float*)d_in[22], (const float*)d_in[26],
                           (const float*)d_in[30], (const float*)d_in[34]};
  const float* out_w = (const float*)d_in[35];
  const float* out_b = (const float*)d_in[36];

  float* ws      = (float*)d_ws;
  float* wo      = ws;                    // 130
  float* midsum  = ws + 130;              // 130
  float* lstsum  = ws + 260;              // 130
  float* midgate = ws + 512;              // 130*128
  float* lstgate = midgate + 16640;       // 130*128
  float* zpageF  = lstgate + 16640;       // 64 f32 = 128 bf16 zero page
  float* accumF  = zpageF + 64;           // 262144 f32 accumulator
  __bf16* bfA    = (__bf16*)(accumF + 262144);   // 262144 bf16 each
  __bf16* bfB    = bfA + 262144;
  __bf16* bfT1   = bfB + 262144;
  __bf16* bfT2   = bfT1 + 262144;
  const __bf16* zpage = (const __bf16*)zpageF;

  route_wo_k   <<<13, 32, 0, stream>>>(alphas, sw, wo);
  route_gates_k<<<130, 128, 0, stream>>>(alphas1, sw1, alphas2, sw2,
                                         lstgate, midgate, lstsum, midsum);
  zero_k<<<1, 64, 0, stream>>>(zpageF, 64);

  // ds0: NCHW f32 x -> NHWC bf16 (2,64,128,16); ds1 -> (2,32,64,64)
  downsampler_k<<<1024, 256, 0, stream>>>(x, nullptr, init_w, init_b, bn0_g, bn0_b, bfA, 3, 13, 128, 256);
  downsampler_k<<<1024, 256, 0, stream>>>(nullptr, bfA, ds1_w, ds1_b, bn1_g, bn1_b, bfB, 16, 48, 64, 128);

  const int DILS[10] = {1, 1, 2, 2, 4, 4, 8, 8, 16, 16};

  __bf16* cur = bfB;   // NHWC (2,32,64,64)
  __bf16* nxt = bfA;
  for (int l = 0; l < 5; ++l) {
    zero_k<<<1024, 256, 0, stream>>>(accumF, 262144);
    for (int i = 0; i < 10; ++i) {
      const int li = l * 10 + i, d = DILS[i];
      const int woff = li * 64 * 64 * 3, boff = li * 64;
      conv3tap_wmma<64, 32, 64><<<128, 256, 0, stream>>>(cur,  g64w[0]+woff, g64b[0]+boff, bfT1, nullptr, nullptr,        nullptr, zpage, 1, 0, 0);
      conv3tap_wmma<64, 32, 64><<<128, 256, 0, stream>>>(bfT1, g64w[1]+woff, g64b[1]+boff, bfT2, nullptr, midgate+li*128, nullptr, zpage, 0, 1, 1);
      conv3tap_wmma<64, 32, 64><<<128, 256, 0, stream>>>(bfT2, g64w[2]+woff, g64b[2]+boff, bfT1, nullptr, nullptr,        nullptr, zpage, d, 0, 0);
      conv3tap_wmma<64, 32, 64><<<128, 256, 0, stream>>>(bfT1, g64w[3]+woff, g64b[3]+boff, nullptr, accumF, lstgate+li*128, wo+li, zpage, 0, d, 2);
    }
    reluconv_k<<<1024, 256, 0, stream>>>(accumF, nxt, 262144);
    __bf16* t = cur; cur = nxt; nxt = t;
  }

  // ds2: NHWC bf16 (2,32,64,64) -> (2,16,32,128)
  downsampler_k<<<512, 256, 0, stream>>>(nullptr, cur, ds2_w, ds2_b, bn2_g, bn2_b, nxt, 64, 64, 32, 64);
  { __bf16* t = cur; cur = nxt; nxt = t; }

  for (int l = 0; l < 8; ++l) {
    const int gl = 5 + l;
    zero_k<<<512, 256, 0, stream>>>(accumF, 131072);
    for (int i = 0; i < 10; ++i) {
      const int li = gl * 10 + i, wi = l * 10 + i, d = DILS[i];
      const int woff = wi * 128 * 128 * 3, boff = wi * 128;
      conv3tap_wmma<128, 16, 32><<<64, 256, 0, stream>>>(cur,  g128w[0]+woff, g128b[0]+boff, bfT1, nullptr, nullptr,        nullptr, zpage, 1, 0, 0);
      conv3tap_wmma<128, 16, 32><<<64, 256, 0, stream>>>(bfT1, g128w[1]+woff, g128b[1]+boff, bfT2, nullptr, midgate+li*128, nullptr, zpage, 0, 1, 1);
      conv3tap_wmma<128, 16, 32><<<64, 256, 0, stream>>>(bfT2, g128w[2]+woff, g128b[2]+boff, bfT1, nullptr, nullptr,        nullptr, zpage, d, 0, 0);
      conv3tap_wmma<128, 16, 32><<<64, 256, 0, stream>>>(bfT1, g128w[3]+woff, g128b[3]+boff, nullptr, accumF, lstgate+li*128, wo+li, zpage, 0, d, 2);
    }
    reluconv_k<<<512, 256, 0, stream>>>(accumF, nxt, 131072);
    __bf16* t = cur; cur = nxt; nxt = t;
  }

  outconv_k<<<80, 256, 0, stream>>>(cur, out_w, out_b, (float*)d_out);
  flops_k  <<<1, 1, 0, stream>>>(wo, midsum, lstsum, (float*)d_out + 20480);
}
